// GreedyDecoder_64020782514176
// MI455X (gfx1250) — matrix operand order repaired and checked by reading, compiled, not run on hardware
//
#include <hip/hip_runtime.h>
#include <hip/hip_bf16.h>
#include <math.h>

// ---------------- problem constants (from reference) ----------------
constexpr int Tt      = 200;
constexpr int Bb      = 128;
constexpr int FH      = 1024;
constexpr int PH      = 320;
constexpr int JHn     = 512;
constexpr int Vv      = 29;
constexpr int BLANKc  = 28;
constexpr int NSTEPS  = 400;
constexpr int MAXLENc = 256;
constexpr int G4  = 4 * PH;    // 1280 (gate width)
constexpr int KC0 = 2 * PH;    // 640  K for [x|h] @ [w_ih;w_hh]
constexpr int KCJ = FH + PH;   // 1344 K for [fi|h1] @ [w_f;w_g]
constexpr int NO  = 32;        // vocab padded to WMMA tile

// ---------------- launch shape ----------------
constexpr int NWG      = 64;
constexpr int NTHREADS = 256;                    // 8 wave32 per WG
constexpr int WPG      = NTHREADS / 32;          // 8 waves per WG
constexpr int NWAVES   = NWG * WPG;              // 512
constexpr int GSTRIDE  = NWG * NTHREADS;         // 16384

typedef _Float16 v16h __attribute__((ext_vector_type(16)));
typedef _Float16 v8h  __attribute__((ext_vector_type(8)));
typedef float    v8f  __attribute__((ext_vector_type(8)));

union FragU { v16h v; v8h h8[2]; };

// ---------------- workspace layout (bytes) ----------------
constexpr size_t al(size_t x){ return (x + 255) & ~size_t(255); }
constexpr size_t OFF_WT0 = 0;                                   // [G4][KC0] f16 (w_ih0;w_hh0)^T
constexpr size_t OFF_WT1 = al(OFF_WT0 + 2ull * G4 * KC0);       // [G4][KC0] f16
constexpr size_t OFF_WTJ = al(OFF_WT1 + 2ull * G4 * KC0);       // [JH][KCJ] f16 (w_f;w_g)^T
constexpr size_t OFF_WTO = al(OFF_WTJ + 2ull * JHn * KCJ);      // [NO][JH]  f16 (w_o)^T padded
constexpr size_t OFF_EMB = al(OFF_WTO + 2ull * NO * JHn);       // [V][PH]   f16
constexpr size_t OFF_A0  = al(OFF_EMB + 2ull * Vv * PH);        // [B][KC0]  f16  [x|h0]
constexpr size_t OFF_A1  = al(OFF_A0  + 2ull * Bb * KC0);       // [B][KC0]  f16  [h0n|h1]
constexpr size_t OFF_AJ  = al(OFF_A1  + 2ull * Bb * KC0);       // [B][KCJ]  f16  [fi|h1n]
constexpr size_t OFF_JH  = al(OFF_AJ  + 2ull * Bb * KCJ);       // [B][JH]   f16  relu joint
constexpr size_t OFF_G0  = al(OFF_JH  + 2ull * Bb * JHn);       // [B][G4]   f32
constexpr size_t OFF_G1  = al(OFF_G0  + 4ull * Bb * G4);        // [B][G4]   f32
constexpr size_t OFF_H0  = al(OFF_G1  + 4ull * Bb * G4);
constexpr size_t OFF_C0  = al(OFF_H0  + 4ull * Bb * PH);
constexpr size_t OFF_H1  = al(OFF_C0  + 4ull * Bb * PH);
constexpr size_t OFF_C1  = al(OFF_H1  + 4ull * Bb * PH);
constexpr size_t OFF_H0N = al(OFF_C1  + 4ull * Bb * PH);
constexpr size_t OFF_C0N = al(OFF_H0N + 4ull * Bb * PH);
constexpr size_t OFF_H1N = al(OFF_C0N + 4ull * Bb * PH);
constexpr size_t OFF_C1N = al(OFF_H1N + 4ull * Bb * PH);
constexpr size_t OFF_BAR = al(OFF_C1N + 4ull * Bb * PH);        // [0]=cnt, [+256B]=gen
constexpr size_t WS_NEED = al(OFF_BAR + 512);                   // ~8 MB

// ---------------- grid-wide barrier (generation counter) ----------------
__device__ __forceinline__ void grid_sync(unsigned* barbase)
{
    unsigned* cnt = barbase;
    unsigned* gen = (unsigned*)((char*)barbase + 256);
    __threadfence();
    __syncthreads();
    if (threadIdx.x == 0) {
        unsigned g = __atomic_load_n(gen, __ATOMIC_ACQUIRE);
        if (atomicAdd(cnt, 1u) == (unsigned)(NWG - 1)) {
            __atomic_store_n(cnt, 0u, __ATOMIC_RELAXED);
            __threadfence();
            __atomic_fetch_add(gen, 1u, __ATOMIC_RELEASE);
        } else {
            while (__atomic_load_n(gen, __ATOMIC_ACQUIRE) == g) {
                __builtin_amdgcn_s_sleep(1);
            }
        }
    }
    __syncthreads();
    __threadfence();
}

__device__ __forceinline__ float sigm(float x) { return 1.0f / (1.0f + __expf(-x)); }

// ---------------- WMMA GEMM helpers ----------------
// A: [128][K] row-major f16. Bt: [N][K] row-major f16 (i.e. B transposed).
// A-frag (16x32 f16, wave32): lane r=l&15,g=l>>4; halves 0-7 = A[r][kk+8g .. +8),
// halves 8-15 = A[r][kk+16+8g .. +8).  B-frag (32x16): lane n=l&15,g=l>>4 holds
// Bt[n][kk+8g .. +8) and Bt[n][kk+16+8g .. +8) (same order as A).
__device__ __forceinline__ v16h load_afrag(const _Float16* __restrict__ Ap, int kk)
{
    FragU f;
    f.h8[0] = *(const v8h*)(Ap + kk);
    f.h8[1] = *(const v8h*)(Ap + kk + 16);
    return f.v;
}

template<int N, int K>
__device__ __forceinline__ void gemm_f32(const _Float16* __restrict__ A,
                                         const _Float16* __restrict__ Bt,
                                         float* __restrict__ C,
                                         int gw, int lane)
{
    constexpr int TILES = (Bb / 16) * (N / 16);
    const int r = lane & 15, g = (lane >> 4) & 1;
    for (int t = gw; t < TILES; t += NWAVES) {
        const int mt = t & 7;
        const int nt = t >> 3;
        const _Float16* Ap = A  + (size_t)(mt * 16 + r) * K + g * 8;
        const _Float16* Bp = Bt + (size_t)(nt * 16 + r) * K + g * 8;
        v8f acc = {};
#pragma unroll
        for (int kk = 0; kk < K; kk += 32) {
            v16h af = load_afrag(Ap, kk);
            v16h bf = load_afrag(Bp, kk);
            acc = __builtin_amdgcn_wmma_f32_16x16x32_f16(false, af, false, bf,
                                                         (short)0, acc, false, false);
        }
        float* Cp = C + (size_t)(mt * 16 + 8 * g) * N + nt * 16 + r;
#pragma unroll
        for (int i = 0; i < 8; i++) Cp[(size_t)i * N] = acc[i];
    }
}

template<int N, int K>
__device__ __forceinline__ void gemm_bias_relu_f16(const _Float16* __restrict__ A,
                                                   const _Float16* __restrict__ Bt,
                                                   _Float16* __restrict__ Cout,
                                                   const float* __restrict__ bias,
                                                   int gw, int lane)
{
    constexpr int TILES = (Bb / 16) * (N / 16);
    const int r = lane & 15, g = (lane >> 4) & 1;
    for (int t = gw; t < TILES; t += NWAVES) {
        const int mt = t & 7;
        const int nt = t >> 3;
        const _Float16* Ap = A  + (size_t)(mt * 16 + r) * K + g * 8;
        const _Float16* Bp = Bt + (size_t)(nt * 16 + r) * K + g * 8;
        v8f acc = {};
#pragma unroll
        for (int kk = 0; kk < K; kk += 32) {
            v16h af = load_afrag(Ap, kk);
            v16h bf = load_afrag(Bp, kk);
            acc = __builtin_amdgcn_wmma_f32_16x16x32_f16(false, af, false, bf,
                                                         (short)0, acc, false, false);
        }
        const int col = nt * 16 + r;
        const float bcol = bias[col];
        _Float16* Cp = Cout + (size_t)(mt * 16 + 8 * g) * N + col;
#pragma unroll
        for (int i = 0; i < 8; i++) {
            float v = acc[i] + bcol;
            Cp[(size_t)i * N] = (_Float16)(v > 0.0f ? v : 0.0f);
        }
    }
}

// Vocab logits, computed redundantly by EVERY workgroup into its own LDS copy
// (so argmax/control can run without an extra grid-wide barrier). WG0 also
// writes the global ys[step] slab.
__device__ __forceinline__ void gemm_vocab_fused(const _Float16* __restrict__ A,
                                                 const _Float16* __restrict__ Bt,
                                                 float* __restrict__ ysp,   // global (WG0 only)
                                                 float* __restrict__ s_y,   // LDS [Bb][NO]
                                                 const float* __restrict__ b_o,
                                                 int wv, int lane, bool writeGlobal)
{
    constexpr int TILES = (Bb / 16) * (NO / 16);   // 16 tiles over 8 waves
    const int r = lane & 15, g = (lane >> 4) & 1;
    for (int t = wv; t < TILES; t += WPG) {
        const int mt = t & 7;
        const int nt = t >> 3;
        const _Float16* Ap = A  + (size_t)(mt * 16 + r) * JHn + g * 8;
        const _Float16* Bp = Bt + (size_t)(nt * 16 + r) * JHn + g * 8;
        v8f acc = {};
#pragma unroll
        for (int kk = 0; kk < JHn; kk += 32) {
            v16h af = load_afrag(Ap, kk);
            v16h bf = load_afrag(Bp, kk);
            acc = __builtin_amdgcn_wmma_f32_16x16x32_f16(false, af, false, bf,
                                                         (short)0, acc, false, false);
        }
        const int col = nt * 16 + r;
        const float bc = (col < Vv) ? b_o[col] : -1.0e30f;
#pragma unroll
        for (int i = 0; i < 8; i++) {
            const int row = mt * 16 + 8 * g + i;
            const float v = acc[i] + bc;
            s_y[row * NO + col] = v;
            if (writeGlobal && col < Vv)
                ysp[(size_t)row * Vv + col] = v;
        }
    }
}

// ---------------- one-time prep: weight transpose+f16, state/output init ----------------
__global__ __launch_bounds__(256) void rnnt_init_kernel(
    const float* __restrict__ embed,
    const float* __restrict__ w_ih0, const float* __restrict__ w_hh0,
    const float* __restrict__ w_ih1, const float* __restrict__ w_hh1,
    const float* __restrict__ w_f,   const float* __restrict__ w_g,
    const float* __restrict__ w_o,
    char* __restrict__ ws, float* __restrict__ out)
{
    const size_t gt = (size_t)blockIdx.x * blockDim.x + threadIdx.x;
    const size_t gs = (size_t)gridDim.x * blockDim.x;
    _Float16* WT0 = (_Float16*)(ws + OFF_WT0);
    _Float16* WT1 = (_Float16*)(ws + OFF_WT1);
    _Float16* WTJ = (_Float16*)(ws + OFF_WTJ);
    _Float16* WTO = (_Float16*)(ws + OFF_WTO);
    _Float16* EMB = (_Float16*)(ws + OFF_EMB);
    float* H0 = (float*)(ws + OFF_H0);
    float* C0 = (float*)(ws + OFF_C0);
    float* H1 = (float*)(ws + OFF_H1);
    float* C1 = (float*)(ws + OFF_C1);

    for (size_t i = gt; i < (size_t)G4 * KC0; i += gs) {
        int n = (int)(i / KC0), k = (int)(i % KC0);
        WT0[i] = (_Float16)((k < PH) ? w_ih0[(size_t)k * G4 + n]
                                     : w_hh0[(size_t)(k - PH) * G4 + n]);
        WT1[i] = (_Float16)((k < PH) ? w_ih1[(size_t)k * G4 + n]
                                     : w_hh1[(size_t)(k - PH) * G4 + n]);
    }
    for (size_t i = gt; i < (size_t)JHn * KCJ; i += gs) {
        int n = (int)(i / KCJ), k = (int)(i % KCJ);
        WTJ[i] = (_Float16)((k < FH) ? w_f[(size_t)k * JHn + n]
                                     : w_g[(size_t)(k - FH) * JHn + n]);
    }
    for (size_t i = gt; i < (size_t)NO * JHn; i += gs) {
        int n = (int)(i / JHn), k = (int)(i % JHn);
        WTO[i] = (n < Vv) ? (_Float16)w_o[(size_t)k * Vv + n] : (_Float16)0.0f;
    }
    for (size_t i = gt; i < (size_t)Vv * PH; i += gs) EMB[i] = (_Float16)embed[i];
    for (size_t i = gt; i < (size_t)Bb * PH; i += gs) {
        H0[i] = 0.0f; C0[i] = 0.0f; H1[i] = 0.0f; C1[i] = 0.0f;
    }
    int* tokens = (int*)(out + (size_t)NSTEPS * Bb * Vv);
    for (size_t i = gt; i < (size_t)Bb * MAXLENc; i += gs) tokens[i] = BLANKc;
    int* olen = tokens + (size_t)Bb * MAXLENc;
    for (size_t i = gt; i < (size_t)Bb; i += gs) olen[i] = 0;
    if (gt == 0) {
        unsigned* bar = (unsigned*)(ws + OFF_BAR);
        bar[0] = 0u;
        *(unsigned*)(ws + OFF_BAR + 256) = 0u;
    }
}

// ---------------- persistent decode kernel: all 400 steps ----------------
__global__ __launch_bounds__(NTHREADS) void rnnt_decode_kernel(
    const float* __restrict__ f, const int* __restrict__ f_lens,
    const float* __restrict__ b0, const float* __restrict__ b1,
    const float* __restrict__ b_j, const float* __restrict__ b_o,
    char* __restrict__ ws, float* __restrict__ out)
{
    const int tid  = threadIdx.x;
    const int wg   = blockIdx.x;
    const int lane = tid & 31;
    const int wv   = tid >> 5;
    const int gw   = wg * WPG + wv;
    const int gt   = wg * NTHREADS + tid;

    const _Float16* WT0 = (const _Float16*)(ws + OFF_WT0);
    const _Float16* WT1 = (const _Float16*)(ws + OFF_WT1);
    const _Float16* WTJ = (const _Float16*)(ws + OFF_WTJ);
    const _Float16* WTO = (const _Float16*)(ws + OFF_WTO);
    const _Float16* EMB = (const _Float16*)(ws + OFF_EMB);
    _Float16* A0  = (_Float16*)(ws + OFF_A0);
    _Float16* A1  = (_Float16*)(ws + OFF_A1);
    _Float16* AJ  = (_Float16*)(ws + OFF_AJ);
    _Float16* JHb = (_Float16*)(ws + OFF_JH);
    float* G0  = (float*)(ws + OFF_G0);
    float* G1  = (float*)(ws + OFF_G1);
    float* H0  = (float*)(ws + OFF_H0);
    float* C0  = (float*)(ws + OFF_C0);
    float* H1  = (float*)(ws + OFF_H1);
    float* C1  = (float*)(ws + OFF_C1);
    float* H0N = (float*)(ws + OFF_H0N);
    float* C0N = (float*)(ws + OFF_C0N);
    float* H1N = (float*)(ws + OFF_H1N);
    float* C1N = (float*)(ws + OFF_C1N);
    unsigned* bar = (unsigned*)(ws + OFF_BAR);

    float* ys   = out;
    int* tokens = (int*)(out + (size_t)NSTEPS * Bb * Vv);
    int* olen   = tokens + (size_t)Bb * MAXLENc;

    // per-WG replicated control state (identical in every WG -> no global races)
    __shared__ float s_y[Bb * NO];                  // 16 KB logits copy
    __shared__ int s_sym[Bb], s_nb[Bb], s_pg[Bb], s_time[Bb];
    __shared__ int s_fin;

    // ---- prologue: state is zero; build A0=[embed[0]|0], AJ fi-part at time 0 ----
    if (tid < Bb) { s_pg[tid] = 0; s_time[tid] = 0; }
    if (tid == 0) s_fin = 0;
    for (int idx = gt; idx < Bb * PH; idx += GSTRIDE) {
        int b = idx / PH, j = idx - b * PH;
        A0[(size_t)b * KC0 + j]      = EMB[j];          // embed[0][j]
        A0[(size_t)b * KC0 + PH + j] = (_Float16)0.0f;  // h0 = 0
    }
    for (int idx = gt; idx < Bb * FH; idx += GSTRIDE) {
        int b = idx / FH, k = idx - b * FH;
        AJ[(size_t)b * KCJ + k] = (_Float16)f[(size_t)b * FH + k];  // f[0][b][k]
    }
    grid_sync(bar);

    for (int step = 0; step < NSTEPS; ++step) {
        // ---- LSTM0 gates: G0 = [x|h0] @ (w_ih0;w_hh0) ----
        gemm_f32<G4, KC0>(A0, WT0, G0, gw, lane);
        grid_sync(bar);

        // ---- cell0 -> h0n,c0n ; build A1=[h0n|h1] ----
        for (int idx = gt; idx < Bb * PH; idx += GSTRIDE) {
            int b = idx / PH, j = idx - b * PH;
            const float* gr = G0 + (size_t)b * G4;
            float gi = gr[j]          + b0[j];
            float gf = gr[PH + j]     + b0[PH + j];
            float gg = gr[2 * PH + j] + b0[2 * PH + j];
            float go = gr[3 * PH + j] + b0[3 * PH + j];
            float cn = sigm(gf) * C0[idx] + sigm(gi) * tanhf(gg);
            float hn = sigm(go) * tanhf(cn);
            C0N[idx] = cn; H0N[idx] = hn;
            A1[(size_t)b * KC0 + j]      = (_Float16)hn;
            A1[(size_t)b * KC0 + PH + j] = (_Float16)H1[idx];
        }
        grid_sync(bar);

        // ---- LSTM1 gates ----
        gemm_f32<G4, KC0>(A1, WT1, G1, gw, lane);
        grid_sync(bar);

        // ---- cell1 -> h1n,c1n ; fill AJ h1n-part ----
        for (int idx = gt; idx < Bb * PH; idx += GSTRIDE) {
            int b = idx / PH, j = idx - b * PH;
            const float* gr = G1 + (size_t)b * G4;
            float gi = gr[j]          + b1[j];
            float gf = gr[PH + j]     + b1[PH + j];
            float gg = gr[2 * PH + j] + b1[2 * PH + j];
            float go = gr[3 * PH + j] + b1[3 * PH + j];
            float cn = sigm(gf) * C1[idx] + sigm(gi) * tanhf(gg);
            float hn = sigm(go) * tanhf(cn);
            C1N[idx] = cn; H1N[idx] = hn;
            AJ[(size_t)b * KCJ + FH + j] = (_Float16)hn;
        }
        grid_sync(bar);

        // ---- joint hidden: relu([fi|h1n] @ (w_f;w_g) + b_j) -> f16 ----
        gemm_bias_relu_f16<JHn, KCJ>(AJ, WTJ, JHb, b_j, gw, lane);
        grid_sync(bar);

        // ---- fused phase: vocab logits (per-WG LDS) + control + commit + prep ----
        float* ysp = ys + (size_t)step * Bb * Vv;
        gemm_vocab_fused(JHb, WTO, ysp, s_y, b_o, wv, lane, wg == 0);
        __syncthreads();

        int fin_old = s_fin;
        if (tid < Bb) {
            int b = tid;
            const float* yr = s_y + b * NO;
            float best = yr[0]; int bi = 0;
            for (int v2 = 1; v2 < Vv; v2++) {
                float vv = yr[v2];
                if (vv > best) { best = vv; bi = v2; }
            }
            int act = fin_old ? 0 : 1;
            int nb  = (bi != BLANKc) & act;
            int bl  = (bi == BLANKc) & act;
            int eos = f_lens[b] - 1;
            int tn  = s_time[b] + bl;
            if (tn > eos) tn = eos;
            s_sym[b]  = bi;
            s_nb[b]   = nb;
            s_pg[b]   = nb ? bi : s_pg[b];
            s_time[b] = tn;
        }
        __syncthreads();
        if (tid == 0) {
            int all = 1;
            for (int b = 0; b < Bb; b++)
                if (s_time[b] != f_lens[b] - 1) { all = 0; break; }
            s_fin = fin_old | all;
        }
        __syncthreads();
        if (wg == 0) {                       // only WG0 touches output control arrays
            if (tid < Bb) {
                int b = tid;
                if (s_nb[b]) {
                    int ol  = olen[b];
                    int pos = ol < MAXLENc - 1 ? ol : MAXLENc - 1;
                    tokens[b * MAXLENc + pos] = s_sym[b];
                    olen[b] = ol + 1;
                }
            }
        }
        // ---- commit state where nb, and build next A0=[embed[pre_g]|h0] ----
        for (int idx = gt; idx < Bb * PH; idx += GSTRIDE) {
            int b = idx / PH, j = idx - b * PH;
            float h0v;
            if (s_nb[b]) {
                h0v = H0N[idx];
                H0[idx] = h0v;      C0[idx] = C0N[idx];
                H1[idx] = H1N[idx]; C1[idx] = C1N[idx];
            } else {
                h0v = H0[idx];
            }
            A0[(size_t)b * KC0 + j]      = EMB[(size_t)s_pg[b] * PH + j];
            A0[(size_t)b * KC0 + PH + j] = (_Float16)h0v;
        }
        // ---- gather fi for next step at updated time ----
        for (int idx = gt; idx < Bb * FH; idx += GSTRIDE) {
            int b = idx / FH, k = idx - b * FH;
            int tb = s_time[b];
            AJ[(size_t)b * KCJ + k] = (_Float16)f[((size_t)tb * Bb + b) * FH + k];
        }
        grid_sync(bar);
    }
}

// ---------------- host entry ----------------
extern "C" void kernel_launch(void* const* d_in, const int* in_sizes, int n_in,
                              void* d_out, int out_size, void* d_ws, size_t ws_size,
                              hipStream_t stream)
{
    const float* f      = (const float*)d_in[0];
    const int*   f_lens = (const int*)  d_in[1];
    const float* embed  = (const float*)d_in[2];
    const float* w_ih0  = (const float*)d_in[3];
    const float* w_hh0  = (const float*)d_in[4];
    const float* b0     = (const float*)d_in[5];
    const float* w_ih1  = (const float*)d_in[6];
    const float* w_hh1  = (const float*)d_in[7];
    const float* b1     = (const float*)d_in[8];
    const float* w_f    = (const float*)d_in[9];
    const float* w_g    = (const float*)d_in[10];
    const float* b_j    = (const float*)d_in[11];
    const float* w_o    = (const float*)d_in[12];
    const float* b_o    = (const float*)d_in[13];
    char*  ws  = (char*)d_ws;
    float* out = (float*)d_out;

    rnnt_init_kernel<<<512, 256, 0, stream>>>(embed, w_ih0, w_hh0, w_ih1, w_hh1,
                                              w_f, w_g, w_o, ws, out);
    rnnt_decode_kernel<<<NWG, NTHREADS, 0, stream>>>(f, f_lens, b0, b1, b_j, b_o,
                                                     ws, out);
    (void)in_sizes; (void)n_in; (void)out_size; (void)ws_size; (void)WS_NEED; (void)Tt;
}
